// MHA_8641474199559
// MI455X (gfx1250) — matrix operand order repaired and checked by reading, not compiled, not run on hardware
//
#include <hip/hip_runtime.h>
#include <hip/hip_bf16.h>
#include <stdint.h>

#define D_MODEL 2048
#define NH      16
#define DH      128
#define B_      2
#define S_      1024
#define SC      1024
#define SF      2048   // SC + S_

typedef __attribute__((ext_vector_type(16))) __bf16 v16bf;
typedef __attribute__((ext_vector_type(8)))  float  v8f;
typedef unsigned short ushort_t;

union FragU { uint4 u[2]; v16bf v; };

__device__ __forceinline__ ushort_t f2bf(float f) {
    union { float f; unsigned u; } c; c.f = f;
    unsigned r = c.u + 0x7FFFu + ((c.u >> 16) & 1u);  // round-to-nearest-even
    return (ushort_t)(r >> 16);
}

#define WMMA_BF16(A, Bm, C) \
    __builtin_amdgcn_wmma_f32_16x16x32_bf16(false, (A), false, (Bm), (short)0, (C), false, false)

// ---------------------------------------------------------------------------
// Elementwise f32 -> bf16 conversion
// ---------------------------------------------------------------------------
__global__ __launch_bounds__(256) void cvt_bf16_kern(const float* __restrict__ src,
                                                     ushort_t* __restrict__ dst, int n) {
    int i = blockIdx.x * 256 + threadIdx.x;
    if (i < n) dst[i] = f2bf(src[i]);
}

// ---------------------------------------------------------------------------
// Copy k_cache/v_cache into k_full/v_full (f32 out) + bf16 K row-major and
// V d-major (transposed) working copies.
// ---------------------------------------------------------------------------
__global__ __launch_bounds__(256) void cache_init_kern(const float* __restrict__ kc,
                                                       const float* __restrict__ vc,
                                                       float* __restrict__ kfull,
                                                       float* __restrict__ vfull,
                                                       ushort_t* __restrict__ Kb,
                                                       ushort_t* __restrict__ Vt) {
    long i = (long)blockIdx.x * 256 + threadIdx.x;
    const long N = (long)B_ * NH * SC * DH;
    if (i >= N) return;
    int d  = (int)(i % DH);
    long t = i / DH;
    int j  = (int)(t % SC);
    int bh = (int)(t / SC);
    float kv = kc[i], vv = vc[i];
    long row = (long)bh * SF + j;
    kfull[row * DH + d] = kv;
    vfull[row * DH + d] = vv;
    Kb[row * DH + d] = f2bf(kv);
    Vt[((long)bh * DH + d) * SF + j] = f2bf(vv);
}

// ---------------------------------------------------------------------------
// Scatter fused-QKV f32 -> head-major bf16 Q, append K/V (f32 to outputs,
// bf16 to working copies; V transposed to d-major).
// ---------------------------------------------------------------------------
__global__ __launch_bounds__(256) void qkv_scatter_kern(const float* __restrict__ QKV,
                                                        ushort_t* __restrict__ Qb,
                                                        float* __restrict__ kfull,
                                                        float* __restrict__ vfull,
                                                        ushort_t* __restrict__ Kb,
                                                        ushort_t* __restrict__ Vt) {
    long i = (long)blockIdx.x * 256 + threadIdx.x;
    const long N = (long)B_ * S_ * 3 * D_MODEL;
    if (i >= N) return;
    int e  = (int)(i % (3 * D_MODEL));
    long t = i / (3 * D_MODEL);
    int s  = (int)(t % S_);
    int b  = (int)(t / S_);
    float v = QKV[i];
    int part = e / D_MODEL;
    int ee   = e % D_MODEL;
    int h = ee / DH, d = ee % DH;
    int bh = b * NH + h;
    if (part == 0) {
        Qb[((long)bh * S_ + s) * DH + d] = f2bf(v);
    } else if (part == 1) {
        long r = (long)bh * SF + (SC + s);
        kfull[r * DH + d] = v;
        Kb[r * DH + d]    = f2bf(v);
    } else {
        long r = (long)bh * SF + (SC + s);
        vfull[r * DH + d] = v;
        Vt[((long)bh * DH + d) * SF + (SC + s)] = f2bf(v);
    }
}

// ---------------------------------------------------------------------------
// GEMM: C[M,N] (f32, row-major) = A[M,K] (bf16 row-major) * W[N,K]^T (bf16).
// 8 waves/block arranged 2(M) x 4(N); each wave owns a 64x64 tile:
// 4 A-frags x 4 B-frags -> 16 WMMA per 32-deep K step (32 FLOP/byte reuse).
// ---------------------------------------------------------------------------
__global__ __launch_bounds__(256) void gemm_bf16_kern(const ushort_t* __restrict__ A,
                                                      const ushort_t* __restrict__ W,
                                                      float* __restrict__ C,
                                                      int M, int N, int K) {
    const int lane = threadIdx.x & 31;
    const int wave = threadIdx.x >> 5;
    const int lm = lane & 15, half = lane >> 4;
    const int m0 = blockIdx.y * 128 + (wave & 1) * 64;
    const int n0 = blockIdx.x * 256 + (wave >> 1) * 64;

    const ushort_t* Ar[4];
    const ushort_t* Wr[4];
#pragma unroll
    for (int t = 0; t < 4; ++t) {
        Ar[t] = A + (long)(m0 + t * 16 + lm) * K;
        Wr[t] = W + (long)(n0 + t * 16 + lm) * K;
    }
    const int aoff = half * 16;  // elements: A halves are contiguous 16-K runs
    const int boff = half * 16;  // elements: B halves are contiguous 16-K runs

    v8f acc[4][4];
#pragma unroll
    for (int i = 0; i < 4; ++i)
#pragma unroll
        for (int j = 0; j < 4; ++j) acc[i][j] = (v8f){};

    for (int k = 0; k < K; k += 32) {
        FragU a[4], b[4];
#pragma unroll
        for (int t = 0; t < 4; ++t) {
            const uint4* pa = (const uint4*)(Ar[t] + k + aoff);
            a[t].u[0] = pa[0]; a[t].u[1] = pa[1];
            const uint4* pb = (const uint4*)(Wr[t] + k + boff);
            b[t].u[0] = pb[0]; b[t].u[1] = pb[1];
        }
#pragma unroll
        for (int i = 0; i < 4; ++i)
#pragma unroll
            for (int j = 0; j < 4; ++j)
                acc[i][j] = WMMA_BF16(a[i].v, b[j].v, acc[i][j]);
    }

#pragma unroll
    for (int i = 0; i < 4; ++i) {
#pragma unroll
        for (int j = 0; j < 4; ++j) {
#pragma unroll
            for (int v = 0; v < 8; ++v) {
                int r = m0 + i * 16 + v + 8 * half;
                C[(long)r * N + n0 + j * 16 + lm] = acc[i][j][v];
            }
        }
    }
}

// ---------------------------------------------------------------------------
// Flash attention: one wave per (b,h, 16-row q-tile). 32-key chunks:
// 8 WMMA scores + online softmax (16-lane shuffles) + P via LDS + 8 WMMA ctx.
// ---------------------------------------------------------------------------
__global__ __launch_bounds__(256) void attn_kern(const ushort_t* __restrict__ Qb,
                                                 const ushort_t* __restrict__ Kb,
                                                 const ushort_t* __restrict__ Vt,
                                                 ushort_t* __restrict__ ctxb) {
    __shared__ ushort_t Pst[8 * 16 * 32];     // 1 KB per wave
    const int lane = threadIdx.x & 31;
    const int wave = threadIdx.x >> 5;
    const int lm = lane & 15, half = lane >> 4;
    const int bh = blockIdx.x;
    const int qt = blockIdx.y * 8 + wave;
    const int b = bh / NH, h = bh % NH;
    ushort_t* Pw = Pst + wave * 512;
    const float scale = 0.08838834764831845f;  // 1/sqrt(128)

    // Q fragments for this 16-row tile (dh = 128 -> 4 K-steps of 32)
    v16bf aq[4];
    const ushort_t* Qrow = Qb + ((long)bh * S_ + qt * 16 + lm) * DH;
#pragma unroll
    for (int kk = 0; kk < 4; ++kk) {
        FragU f;
        const uint4* p = (const uint4*)(Qrow + kk * 32 + half * 16);
        f.u[0] = p[0]; f.u[1] = p[1];
        aq[kk] = f.v;
    }

    const ushort_t* Kbase = Kb + (long)bh * SF * DH;
    const ushort_t* Vbase = Vt + (long)bh * DH * SF;

    v8f O[8]; float mrow[8], lrow[8];
#pragma unroll
    for (int i = 0; i < 8; ++i) { O[i] = (v8f){}; mrow[i] = -INFINITY; lrow[i] = 0.0f; }

    const int kmax = qt * 16 + 15 + SC + 1;   // exclusive key bound for this tile
    for (int k0 = 0; k0 < kmax; k0 += 32) {
        // ---- scores: two 16x16 tiles over 32 keys ----
        v8f s0 = {}, s1 = {};
#pragma unroll
        for (int kk = 0; kk < 4; ++kk) {
            FragU bk0, bk1;
            const uint4* p0 = (const uint4*)(Kbase + (long)(k0 + lm)      * DH + kk * 32 + half * 16);
            const uint4* p1 = (const uint4*)(Kbase + (long)(k0 + 16 + lm) * DH + kk * 32 + half * 16);
            bk0.u[0] = p0[0]; bk0.u[1] = p0[1];
            bk1.u[0] = p1[0]; bk1.u[1] = p1[1];
            s0 = WMMA_BF16(aq[kk], bk0.v, s0);
            s1 = WMMA_BF16(aq[kk], bk1.v, s1);
        }

        // ---- online softmax ----
        const int col0 = k0 + lm, col1 = k0 + 16 + lm;
        float cf[8];
#pragma unroll
        for (int i = 0; i < 8; ++i) {
            const int row = qt * 16 + i + 8 * half;
            float x0 = s0[i] * scale;
            float x1 = s1[i] * scale;
            if (col0 > row + SC) x0 = -3.0e38f;
            if (col1 > row + SC) x1 = -3.0e38f;
            float rm = fmaxf(x0, x1);
            rm = fmaxf(rm, __shfl_xor(rm, 1));
            rm = fmaxf(rm, __shfl_xor(rm, 2));
            rm = fmaxf(rm, __shfl_xor(rm, 4));
            rm = fmaxf(rm, __shfl_xor(rm, 8));
            float mnew = fmaxf(mrow[i], rm);
            float c = __expf(mrow[i] - mnew);
            mrow[i] = mnew;
            float p0v = __expf(x0 - mnew);
            float p1v = __expf(x1 - mnew);
            float rs = p0v + p1v;
            rs += __shfl_xor(rs, 1);
            rs += __shfl_xor(rs, 2);
            rs += __shfl_xor(rs, 4);
            rs += __shfl_xor(rs, 8);
            lrow[i] = lrow[i] * c + rs;
            cf[i] = c;
            const int r = i + 8 * half;
            Pw[r * 32 + lm]      = f2bf(p0v);
            Pw[r * 32 + 16 + lm] = f2bf(p1v);
        }
        // per-wave LDS fence (no block barrier: waves have different trip counts)
        asm volatile("s_wait_dscnt 0" ::: "memory");

        // A-fragment of P (16x32) from LDS
        FragU pa;
        const uint4* pp = (const uint4*)(Pw + lm * 32 + half * 16);
        pa.u[0] = pp[0]; pa.u[1] = pp[1];

        // ---- ctx accumulate: 8 output-d chunks of 16 ----
#pragma unroll
        for (int nc = 0; nc < 8; ++nc) {
            FragU bv;
            const uint4* pv = (const uint4*)(Vbase + (long)(nc * 16 + lm) * SF + k0 + half * 16);
            bv.u[0] = pv[0]; bv.u[1] = pv[1];
#pragma unroll
            for (int i = 0; i < 8; ++i) O[nc][i] *= cf[i];
            O[nc] = WMMA_BF16(pa.v, bv.v, O[nc]);
        }
    }

    // ---- normalize and store ctx as bf16 into (B,S,D) for the out-proj ----
#pragma unroll
    for (int nc = 0; nc < 8; ++nc) {
#pragma unroll
        for (int i = 0; i < 8; ++i) {
            const int row = qt * 16 + i + 8 * half;
            float val = O[nc][i] / lrow[i];
            ctxb[((long)(b * S_ + row)) * D_MODEL + h * DH + nc * 16 + lm] = f2bf(val);
        }
    }
}

// ---------------------------------------------------------------------------
// Host-side launcher
// ---------------------------------------------------------------------------
extern "C" void kernel_launch(void* const* d_in, const int* in_sizes, int n_in,
                              void* d_out, int out_size, void* d_ws, size_t ws_size,
                              hipStream_t stream) {
    const float* x    = (const float*)d_in[0];
    const float* kc   = (const float*)d_in[1];
    const float* vc   = (const float*)d_in[2];
    const float* qkv  = (const float*)d_in[3];
    const float* wo   = (const float*)d_in[4];
    (void)in_sizes; (void)n_in; (void)out_size; (void)ws_size;

    float* out   = (float*)d_out;                       // (B,S,D)
    float* kfull = out   + (long)B_ * S_ * D_MODEL;     // (B,H,SF,DH)
    float* vfull = kfull + (long)B_ * NH * SF * DH;     // (B,H,SF,DH)

    char* ws = (char*)d_ws;
    ushort_t* Xb    = (ushort_t*)(ws);                                  //  8 MB  x bf16
    ushort_t* Wqkvb = (ushort_t*)(ws + 8388608);                        // 24 MB  qkv bf16
    ushort_t* Wob   = (ushort_t*)(ws + 33554432);                       //  8 MB  wo bf16
    float*    QKVf  = (float*)   (ws + 41943040);                       // 48 MB  fused QKV f32
    ushort_t* Qbh   = (ushort_t*)(ws + 92274688);                       //  8 MB  Q head-major bf16
    ushort_t* Kbf   = (ushort_t*)(ws + 100663296);                      // 16 MB  K row-major bf16
    ushort_t* Vtf   = (ushort_t*)(ws + 117440512);                      // 16 MB  V d-major bf16
    ushort_t* Ctxb  = (ushort_t*)(ws + 134217728);                      //  8 MB  ctx bf16

    const int nX  = B_ * S_ * D_MODEL;          // 4194304
    const int nW  = 3 * D_MODEL * D_MODEL;      // 12582912
    const int nWo = D_MODEL * D_MODEL;          // 4194304
    const long nCache = (long)B_ * NH * SC * DH;    // 4194304
    const long nQKV   = (long)B_ * S_ * 3 * D_MODEL; // 12582912

    cvt_bf16_kern<<<(nX  + 255) / 256, 256, 0, stream>>>(x,   Xb,    nX);
    cvt_bf16_kern<<<(nW  + 255) / 256, 256, 0, stream>>>(qkv, Wqkvb, nW);
    cvt_bf16_kern<<<(nWo + 255) / 256, 256, 0, stream>>>(wo,  Wob,   nWo);
    cache_init_kern<<<(unsigned)((nCache + 255) / 256), 256, 0, stream>>>(
        kc, vc, kfull, vfull, Kbf, Vtf);

    // QKV projection: (B*S, 2048) x (6144, 2048)^T -> (B*S, 6144)
    // block tile 128x256 -> grid (6144/256, 2048/128)
    gemm_bf16_kern<<<dim3(6144 / 256, 2048 / 128), 256, 0, stream>>>(
        Xb, Wqkvb, QKVf, B_ * S_, 3 * D_MODEL, D_MODEL);

    qkv_scatter_kern<<<(unsigned)((nQKV + 255) / 256), 256, 0, stream>>>(
        QKVf, Qbh, kfull, vfull, Kbf, Vtf);

    // Attention: 32 (b,h) x 8 blocks of 8 wave-tiles
    attn_kern<<<dim3(B_ * NH, 8), 256, 0, stream>>>(Qbh, Kbf, Vtf, Ctxb);

    // Output projection: (B*S, 2048) x (2048, 2048)^T -> out
    gemm_bf16_kern<<<dim3(2048 / 256, 2048 / 128), 256, 0, stream>>>(
        Ctxb, Wob, out, B_ * S_, D_MODEL, D_MODEL);
}
